// PSABlock_13486197310329
// MI455X (gfx1250) — matrix operand order repaired and checked by reading, not compiled, not run on hardware
//
#include <hip/hip_runtime.h>

typedef __attribute__((ext_vector_type(16))) _Float16 v16h;
typedef __attribute__((ext_vector_type(8)))  _Float16 v8h;
typedef __attribute__((ext_vector_type(4)))  _Float16 v4h;
typedef __attribute__((ext_vector_type(8)))  float    v8f;

#define BATCH 16
#define CCH   256
#define NTOK  4096
#define TM 128
#define TN 64
#define TK 32
#define LDA 48   // halves per A row in LDS (96B rows: 16B-aligned, bank-staggered)
#define LDB 48   // halves per B row in LDS

// ---- fp32->f16 / f16 pass-through loaders for the X (K x N) operand ----
__device__ __forceinline__ void load_x8(const float* __restrict__ s, _Float16 o[8]) {
  float4 u0 = *(const float4*)s;
  float4 u1 = *(const float4*)(s + 4);
  o[0]=(_Float16)u0.x; o[1]=(_Float16)u0.y; o[2]=(_Float16)u0.z; o[3]=(_Float16)u0.w;
  o[4]=(_Float16)u1.x; o[5]=(_Float16)u1.y; o[6]=(_Float16)u1.z; o[7]=(_Float16)u1.w;
}
__device__ __forceinline__ void load_x8(const _Float16* __restrict__ s, _Float16 o[8]) {
  v8h u = *(const v8h*)s;
#pragma unroll
  for (int i = 0; i < 8; ++i) o[i] = u[i];
}

__device__ __forceinline__ void loadA_g(const float* __restrict__ s, float4 fr[4]) {
#pragma unroll
  for (int i = 0; i < 4; ++i) fr[i] = *(const float4*)(s + i * 4);
}

// MODE 0: OUT(f32) = W@X
// MODE 1: OUT(f32) = W@X + bias(p0) + RES
// MODE 2: OUT(f16) = SiLU(BN(W@X; p0..p3))
// MODE 3: OUT(f32) = BN(W@X; p0..p3) + RES
// Compile-time dims (M,K template; N fixed = NTOK) so the epilogue's 8 row
// stores per fragment fold v*N*4 into global_store immediate offsets.
template<int MODE, typename XT, int M, int K>
__global__ __launch_bounds__(256) void gemm_wmma_kernel(
    const float* __restrict__ Wm,   // [M,K] shared across batch
    const XT*    __restrict__ X,    // [B][K,N]
    void*        __restrict__ OUTv, // [B][M,N] f32 or f16
    const float* __restrict__ RES,  // [B][M,N] residual (modes 1,3)
    const float* __restrict__ p0, const float* __restrict__ p1,
    const float* __restrict__ p2, const float* __restrict__ p3)
{
  __shared__ __align__(16) _Float16 As[TM * LDA];
  __shared__ __align__(16) _Float16 Bs[TN * LDB];

  const int bb = blockIdx.z;
  const int m0 = blockIdx.y * TM;
  const int n0 = blockIdx.x * TN;
  const int t    = threadIdx.x;
  const int lane = t & 31;
  const int wave = t >> 5;
  const int wm = wave & 3;        // 4 waves along M  (32 rows each)
  const int wn = wave >> 2;       // 2 waves along N  (32 cols each)
  const int r  = lane & 15;       // row/col within 16x16 fragment
  const int hh = lane >> 4;       // half-wave select

  const XT* __restrict__ Xb = X + (size_t)bb * K * NTOK;

  // Per-thread cooperative-load coordinates
  const int arow = t >> 1;               // 128 rows, 2 threads/row
  const int aseg = (t & 1) * 16;         // 16 floats each
  const int bkr  = t >> 3;               // 32 k-rows, 8 threads/row
  const int bns  = (t & 7) * 8;          // 8 n-columns each

  const float* __restrict__ aG = Wm + (size_t)(m0 + arow) * K + aseg;
  const XT*    __restrict__ bG = Xb + (size_t)bkr * NTOK + n0 + bns;

  v8f acc[2][2] = {};

  // ---- software pipeline: stage tile k+1 in regs while WMMAs eat tile k ----
  float4   ar[4];
  _Float16 br[8];
  loadA_g(aG, ar);
  load_x8(bG, br);

  for (int k0 = 0; k0 < K; k0 += TK) {
    // commit staged tile to LDS (A row-major f16; B transposed [n][k] f16)
    {
      _Float16* dst = &As[arow * LDA + aseg];
#pragma unroll
      for (int i = 0; i < 4; ++i) {
        v4h hv = { (_Float16)ar[i].x, (_Float16)ar[i].y,
                   (_Float16)ar[i].z, (_Float16)ar[i].w };
        *(v4h*)(dst + i * 4) = hv;
      }
#pragma unroll
      for (int i = 0; i < 8; ++i) Bs[(bns + i) * LDB + bkr] = br[i];
    }
    __syncthreads();

    // stage next K-tile (overlaps with fragment loads + WMMAs below)
    if (k0 + TK < K) {
      loadA_g(aG + k0 + TK, ar);
      load_x8(bG + (size_t)(k0 + TK) * NTOK, br);
      if (k0 + 2 * TK < K)
        __builtin_prefetch(aG + k0 + 2 * TK, 0, 3);   // global_prefetch_b8, WGP scope
    }

    // ---- fragments per ISA 7.12.2 lane layouts ----
    v16h a[2], bf[2];
#pragma unroll
    for (int mi = 0; mi < 2; ++mi) {
      const int row = wm * 32 + mi * 16 + r;
      union { v16h v; v8h h2[2]; } u;
      u.h2[0] = *(const v8h*)&As[row * LDA + 8 * hh];        // K = 8h .. 8h+7
      u.h2[1] = *(const v8h*)&As[row * LDA + 16 + 8 * hh];   // K = 16+8h .. 16+8h+7
      a[mi] = u.v;
    }
#pragma unroll
    for (int ni = 0; ni < 2; ++ni) {
      const int col = wn * 32 + ni * 16 + r;
      union { v16h v; v8h h2[2]; } ub;
      ub.h2[0] = *(const v8h*)&Bs[col * LDB + 8 * hh];        // K = 8h .. 8h+7
      ub.h2[1] = *(const v8h*)&Bs[col * LDB + 16 + 8 * hh];   // K = 16+8h .. 16+8h+7
      bf[ni] = ub.v;
    }
#pragma unroll
    for (int mi = 0; mi < 2; ++mi)
#pragma unroll
      for (int ni = 0; ni < 2; ++ni)
        acc[mi][ni] = __builtin_amdgcn_wmma_f32_16x16x32_f16(
            false, a[mi], false, bf[ni], (short)0, acc[mi][ni], false, false);
    __syncthreads();
  }

  // ---- epilogue: C layout = VGPR v -> M = 8*h + v, N = lane%16 ----
  // One base address per fragment; v*NTOK folds into store immediate offsets.
#pragma unroll
  for (int mi = 0; mi < 2; ++mi)
#pragma unroll
    for (int ni = 0; ni < 2; ++ni) {
      const int mb = m0 + wm * 32 + mi * 16 + hh * 8;
      const int n  = n0 + wn * 32 + ni * 16 + r;
      const size_t idx0 = (size_t)bb * (M * NTOK) + (size_t)mb * NTOK + n;
#pragma unroll
      for (int v = 0; v < 8; ++v) {
        const float val = acc[mi][ni][v];
        if (MODE == 0) {
          ((float*)OUTv)[idx0 + v * NTOK] = val;
        } else if (MODE == 1) {
          ((float*)OUTv)[idx0 + v * NTOK] = val + p0[mb + v] + RES[idx0 + v * NTOK];
        } else if (MODE == 2) {
          const float sc = p0[mb + v] * rsqrtf(p3[mb + v] + 1e-5f);
          const float sh = p1[mb + v] - p2[mb + v] * sc;
          const float tv = val * sc + sh;
          const float sw = tv / (1.0f + __expf(-tv));       // SiLU
          ((_Float16*)OUTv)[idx0 + v * NTOK] = (_Float16)sw;
        } else {
          const float sc = p0[mb + v] * rsqrtf(p3[mb + v] + 1e-5f);
          const float sh = p1[mb + v] - p2[mb + v] * sc;
          ((float*)OUTv)[idx0 + v * NTOK] = RES[idx0 + v * NTOK] + val * sc + sh;
        }
      }
    }
}

// ---------------- attention statistics kernels (fp32, bandwidth-bound) ----------------

// inv_nrm[row] = 1 / max(||W[row,:]||, 1e-12), rows = B*C
__global__ __launch_bounds__(256) void rownorm_kernel(const float* __restrict__ Wt,
                                                      float* __restrict__ inv) {
  __shared__ float red[256];
  const int row = blockIdx.x;
  const float* p = Wt + (size_t)row * NTOK;
  float s = 0.0f;
  for (int i = threadIdx.x; i < NTOK; i += 256) { float v = p[i]; s += v * v; }
  red[threadIdx.x] = s; __syncthreads();
  for (int off = 128; off > 0; off >>= 1) {
    if (threadIdx.x < off) red[threadIdx.x] += red[threadIdx.x + off];
    __syncthreads();
  }
  if (threadIdx.x == 0) inv[row] = 1.0f / fmaxf(sqrtf(red[0]), 1e-12f);
}

// Pi[b,h,n] = softmax_h( temp[h] * sum_d (W[b,h,d,n]*inv[b,h,d])^2 )
__global__ __launch_bounds__(256) void pi_kernel(const float* __restrict__ Wt,
                                                 const float* __restrict__ inv,
                                                 const float* __restrict__ temp,
                                                 float* __restrict__ Pi) {
  const int gid = blockIdx.x * 256 + threadIdx.x;     // over B*N
  const int b = gid >> 12, n = gid & (NTOK - 1);
  float s[4];
#pragma unroll
  for (int h = 0; h < 4; ++h) {
    float acc = 0.0f;
    const float* base  = Wt + ((size_t)b * CCH + h * 64) * NTOK + n;
    const float* ibase = inv + b * CCH + h * 64;
    for (int d = 0; d < 64; ++d) {
      float v = base[(size_t)d * NTOK] * ibase[d];
      acc += v * v;
    }
    s[h] = acc * temp[h];
  }
  const float mx = fmaxf(fmaxf(s[0], s[1]), fmaxf(s[2], s[3]));
  float e[4], sum = 0.0f;
#pragma unroll
  for (int h = 0; h < 4; ++h) { e[h] = __expf(s[h] - mx); sum += e[h]; }
  const float rs = 1.0f / sum;
#pragma unroll
  for (int h = 0; h < 4; ++h) Pi[((size_t)(b * 4 + h)) * NTOK + n] = e[h] * rs;
}

// sumPi[row] = sum_n Pi[row,n], rows = B*HEADS
__global__ __launch_bounds__(256) void rowsum_kernel(const float* __restrict__ Pi,
                                                     float* __restrict__ out) {
  __shared__ float red[256];
  const int row = blockIdx.x;
  const float* p = Pi + (size_t)row * NTOK;
  float s = 0.0f;
  for (int i = threadIdx.x; i < NTOK; i += 256) s += p[i];
  red[threadIdx.x] = s; __syncthreads();
  for (int off = 128; off > 0; off >>= 1) {
    if (threadIdx.x < off) red[threadIdx.x] += red[threadIdx.x + off];
    __syncthreads();
  }
  if (threadIdx.x == 0) out[row] = red[0];
}

// attn[b,c] = 1 / (1 + (sum_n Pi * W^2) / (sumPi + 1e-8))
__global__ __launch_bounds__(256) void dots_kernel(const float* __restrict__ Wt,
                                                   const float* __restrict__ Pi,
                                                   const float* __restrict__ sumPi,
                                                   float* __restrict__ attn) {
  __shared__ float red[256];
  const int row = blockIdx.x;                 // b*C + c
  const int b = row >> 8, c = row & 255, h = c >> 6;
  const float* wp = Wt + (size_t)row * NTOK;
  const float* pp = Pi + ((size_t)(b * 4 + h)) * NTOK;
  float s = 0.0f;
  for (int i = threadIdx.x; i < NTOK; i += 256) { float w = wp[i]; s += pp[i] * w * w; }
  red[threadIdx.x] = s; __syncthreads();
  for (int off = 128; off > 0; off >>= 1) {
    if (threadIdx.x < off) red[threadIdx.x] += red[threadIdx.x + off];
    __syncthreads();
  }
  if (threadIdx.x == 0) {
    const float dots = red[0] / (sumPi[b * 4 + h] + 1e-8f);
    attn[row] = 1.0f / (1.0f + dots);
  }
}

// W[b,c,n] = -(W * Pi[b,h,n]) * attn[b,c], in place (float4 vectorized)
__global__ __launch_bounds__(256) void scale_kernel(float* __restrict__ Wt,
                                                    const float* __restrict__ Pi,
                                                    const float* __restrict__ attn) {
  const size_t gid  = (size_t)blockIdx.x * 256 + threadIdx.x;
  const size_t idx4 = gid * 4;
  const size_t row  = idx4 / NTOK;
  const int    n    = (int)(idx4 & (NTOK - 1));
  const int b = (int)(row >> 8), c = (int)(row & 255), h = c >> 6;
  const float a = -attn[row];
  const float4 pv = *(const float4*)&Pi[((size_t)(b * 4 + h)) * NTOK + n];
  float4 wv = *(float4*)&Wt[idx4];
  wv.x = wv.x * pv.x * a;
  wv.y = wv.y * pv.y * a;
  wv.z = wv.z * pv.z * a;
  wv.w = wv.w * pv.w * a;
  *(float4*)&Wt[idx4] = wv;
}

extern "C" void kernel_launch(void* const* d_in, const int* in_sizes, int n_in,
                              void* d_out, int out_size, void* d_ws, size_t ws_size,
                              hipStream_t stream) {
  const float* x       = (const float*)d_in[0];
  const float* qkv_w   = (const float*)d_in[1];
  const float* temp    = (const float*)d_in[2];
  const float* proj_w  = (const float*)d_in[3];
  const float* proj_b  = (const float*)d_in[4];
  const float* conv1_w = (const float*)d_in[5];
  const float* bn1_g   = (const float*)d_in[6];
  const float* bn1_b   = (const float*)d_in[7];
  const float* bn1_m   = (const float*)d_in[8];
  const float* bn1_v   = (const float*)d_in[9];
  const float* conv2_w = (const float*)d_in[10];
  const float* bn2_g   = (const float*)d_in[11];
  const float* bn2_b   = (const float*)d_in[12];
  const float* bn2_m   = (const float*)d_in[13];
  const float* bn2_v   = (const float*)d_in[14];
  float* out = (float*)d_out;

  // Workspace layout (~65.1 MiB):
  //   Wt   : B*C*N fp32  (qkv result -> attention out in place; later aliased as f16 FFN hidden)
  //   Pi   : B*H*N fp32
  //   inv  : B*C fp32 ; attn : B*C fp32 ; sumPi : B*H fp32
  char* ws = (char*)d_ws;
  const size_t szW = (size_t)BATCH * CCH * NTOK * sizeof(float);   // 64 MiB
  float* Wt    = (float*)ws;
  float* Pi    = (float*)(ws + szW);
  float* invn  = (float*)(ws + szW + (size_t)BATCH * 4 * NTOK * sizeof(float));
  float* attn  = invn + BATCH * CCH;
  float* sumPi = attn + BATCH * CCH;

  const dim3 blk(256);
  const dim3 gA(NTOK / TN, CCH / TM, BATCH);        // (64, 2, 16)
  const dim3 gC1(NTOK / TN, 2 * CCH / TM, BATCH);   // (64, 4, 16)

  // 1) W = qkv_w @ x_b                (per-batch [256x256]@[256x4096])
  gemm_wmma_kernel<0, float, CCH, CCH><<<gA, blk, 0, stream>>>(
      qkv_w, x, Wt, nullptr, nullptr, nullptr, nullptr, nullptr);
  // 2) attention statistics
  rownorm_kernel<<<BATCH * CCH, blk, 0, stream>>>(Wt, invn);
  pi_kernel<<<BATCH * NTOK / 256, blk, 0, stream>>>(Wt, invn, temp, Pi);
  rowsum_kernel<<<BATCH * 4, blk, 0, stream>>>(Pi, sumPi);
  dots_kernel<<<BATCH * CCH, blk, 0, stream>>>(Wt, Pi, sumPi, attn);
  scale_kernel<<<(int)((size_t)BATCH * CCH * NTOK / 4 / 256), blk, 0, stream>>>(Wt, Pi, attn);
  // 3) xr = x + proj_w @ out + proj_b   -> d_out (fp32 residual carrier)
  gemm_wmma_kernel<1, float, CCH, CCH><<<gA, blk, 0, stream>>>(
      proj_w, Wt, out, x, proj_b, nullptr, nullptr, nullptr);
  // 4) y = SiLU(BN1(conv1_w @ xr))      -> f16, aliases Wt region (64 MiB)
  gemm_wmma_kernel<2, float, 2 * CCH, CCH><<<gC1, blk, 0, stream>>>(
      conv1_w, out, (void*)Wt, nullptr, bn1_g, bn1_b, bn1_m, bn1_v);
  // 5) out = xr + BN2(conv2_w @ y)      (f16 X operand), element-exact overwrite of d_out
  gemm_wmma_kernel<3, _Float16, CCH, 2 * CCH><<<gA, blk, 0, stream>>>(
      conv2_w, (const _Float16*)Wt, out, out, bn2_g, bn2_b, bn2_m, bn2_v);
}
